// QStack_15788299780328
// MI455X (gfx1250) — matrix operand order repaired and checked by reading, not compiled, not run on hardware
//
#include <hip/hip_runtime.h>

// ---------------------------------------------------------------------------
// Types for CDNA5 WMMA (wave32): bf16 16x16x32, f32 accumulate.
// ---------------------------------------------------------------------------
typedef __attribute__((ext_vector_type(16))) __bf16 v16bf;
typedef __attribute__((ext_vector_type(8)))  __bf16 v8bf;
typedef __attribute__((ext_vector_type(8)))  float  v8f;

__device__ __forceinline__ unsigned short f2bf(float f) {
  unsigned u = __float_as_uint(f);
  u += 0x7FFFu + ((u >> 16) & 1u);          // round-to-nearest-even
  return (unsigned short)(u >> 16);
}

__device__ __forceinline__ v8bf zero8bf() {
  union { v8bf v; unsigned u[4]; } z;
  z.u[0] = z.u[1] = z.u[2] = z.u[3] = 0u;
  return z.v;
}

__device__ __forceinline__ v8bf ld8(const unsigned short* p) {
  return *(const v8bf*)p;
}

__device__ __forceinline__ v16bf mk16(v8bf lo, v8bf hb) {
  union { v16bf v; v8bf h[2]; } u;
  u.h[0] = lo; u.h[1] = hb;
  return u.v;
}

// ---------------------------------------------------------------------------
// Weight / input packing kernels (f32 -> bf16, layout transforms)
// ---------------------------------------------------------------------------

// enc0_w (256,3,4,4) -> Wp0 [256][64] (K = ic*16+kh*4+kw, padded 48..63 = 0)
__global__ void pack_wenc0_k(const float* __restrict__ w, unsigned short* __restrict__ wp) {
  int t = blockIdx.x * 256 + threadIdx.x;          // 256*64
  int oc = t >> 6, k = t & 63;
  float v = (k < 48) ? w[oc * 48 + k] : 0.f;
  wp[t] = f2bf(v);
}

// x (32,3,128,128) -> im2col A0 [131072][64], stride-2 conv, pad 1, k=4
__global__ void pack_im2col0_k(const float* __restrict__ x, unsigned short* __restrict__ a0) {
  int t = blockIdx.x * 256 + threadIdx.x;          // 131072*64
  int row = t >> 6, k = t & 63;
  float v = 0.f;
  if (k < 48) {
    int b = row >> 12, oh = (row >> 6) & 63, ow = row & 63;
    int ic = k >> 4, kh = (k >> 2) & 3, kw = k & 3;
    int ih = oh * 2 - 1 + kh, iw = ow * 2 - 1 + kw;
    if ((unsigned)ih < 128u && (unsigned)iw < 128u)
      v = x[(((size_t)(b * 3 + ic)) * 128 + ih) * 128 + iw];
  }
  a0[t] = f2bf(v);
}

// cb0,cb1 (2,512,128) -> cbp [stage][g][512][128] bf16 + cnorm [stage*2+g][512]
__global__ void pack_cb_k(const float* __restrict__ cb0, const float* __restrict__ cb1,
                          unsigned short* __restrict__ cbp, float* __restrict__ cnorm) {
  int t = blockIdx.x * 256 + threadIdx.x;          // 2048
  int stage = t >> 10, rem = t & 1023;
  const float* src = (stage ? cb1 : cb0) + (size_t)rem * 128;
  unsigned short* dst = cbp + (size_t)t * 128;
  float s = 0.f;
  for (int i = 0; i < 128; ++i) { float v = src[i]; s += v * v; dst[i] = f2bf(v); }
  cnorm[t] = s;
}

// enc1_w (256,256,4,4) -> Wp1 [256][4096], K = (kh*4+kw)*256 + ic (tap-major)
__global__ void pack_wenc1_k(const float* __restrict__ w, unsigned short* __restrict__ wp) {
  int t = blockIdx.x * 256 + threadIdx.x;          // 256*4096
  int oc = t >> 12, k = t & 4095;
  int tap = k >> 8, ic = k & 255;
  int kh = tap >> 2, kw = tap & 3;
  wp[t] = f2bf(w[(((size_t)(oc * 256 + ic)) * 4 + kh) * 4 + kw]);
}

// dec1_w HWIO (4,4,256,256) -> Wp1d [4 parity classes][256][1024]
// class pc=(ph,pw): tap t -> (kh,kw) = (ph+2*(t>>1), pw+2*(t&1)); K = t*256+ic
__global__ void pack_wdec1_k(const float* __restrict__ w, unsigned short* __restrict__ wp) {
  int t = blockIdx.x * 256 + threadIdx.x;          // 4*256*1024
  int pc = t >> 18, rem = t & ((1 << 18) - 1);
  int oc = rem >> 10, k = rem & 1023;
  int tp = k >> 8, ic = k & 255;
  int kh = (pc >> 1) + 2 * (tp >> 1), kw = (pc & 1) + 2 * (tp & 1);
  wp[t] = f2bf(w[(((size_t)(kh * 4 + kw)) * 256 + ic) * 256 + oc]);
}

// dec0_w HWIO (4,4,256,3) -> Wp0d [4][16][1024] (oc padded 3..15 = 0)
__global__ void pack_wdec0_k(const float* __restrict__ w, unsigned short* __restrict__ wp) {
  int t = blockIdx.x * 256 + threadIdx.x;          // 4*16*1024
  int pc = t >> 14, rem = t & 16383;
  int oc = rem >> 10, k = rem & 1023;
  int tp = k >> 8, ic = k & 255;
  int kh = (pc >> 1) + 2 * (tp >> 1), kw = (pc & 1) + 2 * (tp & 1);
  float v = (oc < 3) ? w[(((size_t)(kh * 4 + kw)) * 256 + ic) * 3 + oc] : 0.f;
  wp[t] = f2bf(v);
}

// ---------------------------------------------------------------------------
// Implicit-GEMM WMMA kernel with double-buffered fragment pipeline.
//   AMODE 0: A packed row-major M x K
//   AMODE 1: A = stride-2 conv (pad 1, k=4) over ch-last input (B,64,64,256)
//   AMODE 2: A = conv-transpose parity class (ph,pw) over ch-last (B,IH,IH,256)
//   OMODE 0: bf16 ch-last M x Ntot, + bias
//   OMODE 1: bf16 convT scatter to (B,2IH,2IH,256), + bias
//   OMODE 2: f32 NCHW (B,3,128,128) scatter, n<3 only, + bias
// Block = 256 threads = 8 waves arranged WM x WN; wave tile = (MF*16) x (NF*16).
// ---------------------------------------------------------------------------
template <int AMODE, int OMODE, int WM, int WN, int MF, int NF>
__global__ void __launch_bounds__(256)
gemm_k(const unsigned short* __restrict__ A,
       const unsigned short* __restrict__ W,
       const float* __restrict__ bias,
       void* __restrict__ outp,
       int Ntot, int K, int ph, int pw, int ihshift) {
  const int tid  = threadIdx.x;
  const int wave = tid >> 5;
  const int lane = tid & 31;
  const int lr   = lane & 15;
  const int hi   = lane >> 4;
  const int wm   = wave % WM;
  const int wn   = wave / WM;
  const int m0   = (blockIdx.x * WM + wm) * (MF * 16);
  const int n0   = (blockIdx.y * WN + wn) * (NF * 16);
  const int IH   = 1 << ihshift;

  // per-fragment row decode (lane lr = row within fragment)
  int rb[MF], ry[MF], rx[MF];
#pragma unroll
  for (int f = 0; f < MF; ++f) {
    int r = m0 + f * 16 + lr;
    if constexpr (AMODE == 1) {
      rb[f] = r >> 10; ry[f] = (r >> 5) & 31; rx[f] = r & 31;
    } else if constexpr (AMODE == 2) {
      rb[f] = r >> (2 * ihshift); ry[f] = (r >> ihshift) & (IH - 1); rx[f] = r & (IH - 1);
    } else {
      rb[f] = r; ry[f] = 0; rx[f] = 0;
    }
  }

  auto loadA = [&](int f, int kc) -> v16bf {
    if constexpr (AMODE == 0) {
      const unsigned short* p = A + (size_t)rb[f] * K + kc;
      return mk16(ld8(p + hi * 8), ld8(p + 16 + hi * 8));
    } else {
      const int tap  = kc >> 8;      // 256 channels per tap, 8 chunks each
      const int coff = kc & 255;
      int ih_, iw_, span;
      if constexpr (AMODE == 1) {
        int kh = tap >> 2, kw = tap & 3;
        ih_ = ry[f] * 2 - 1 + kh; iw_ = rx[f] * 2 - 1 + kw; span = 64;
      } else {
        int th = tap >> 1, tw = tap & 1;
        ih_ = ry[f] + ph - 1 + th; iw_ = rx[f] + pw - 1 + tw; span = IH;
      }
      bool valid = (unsigned)ih_ < (unsigned)span && (unsigned)iw_ < (unsigned)span;
      int ihc = valid ? ih_ : 0, iwc = valid ? iw_ : 0;
      const unsigned short* p =
          A + (((size_t)((rb[f] * span + ihc) * span + iwc)) << 8) + coff;
      v8bf lo = ld8(p + hi * 8);
      v8bf hb = ld8(p + 16 + hi * 8);
      if (!valid) { lo = zero8bf(); hb = zero8bf(); }
      return mk16(lo, hb);
    }
  };
  auto loadB = [&](int f, int kc) -> v16bf {
    const unsigned short* p = W + (size_t)(n0 + f * 16 + lr) * K + kc + hi * 8;
    return mk16(ld8(p), ld8(p + 16));
  };

  v8f acc[MF][NF];
#pragma unroll
  for (int a = 0; a < MF; ++a)
#pragma unroll
    for (int b = 0; b < NF; ++b)
#pragma unroll
      for (int i = 0; i < 8; ++i) acc[a][b][i] = 0.f;

  // Double-buffered K pipeline; buffer selection is compile-time (unrolled x2)
  // so fragments stay in VGPRs and next-step loads overlap current WMMAs.
  v16bf afr0[MF], bfr0[NF], afr1[MF], bfr1[NF];
#pragma unroll
  for (int f = 0; f < MF; ++f) afr0[f] = loadA(f, 0);
#pragma unroll
  for (int f = 0; f < NF; ++f) bfr0[f] = loadB(f, 0);

  for (int kc = 0; kc < K; kc += 64) {
    const bool more1 = (kc + 32) < K;
    if (more1) {
#pragma unroll
      for (int f = 0; f < MF; ++f) afr1[f] = loadA(f, kc + 32);
#pragma unroll
      for (int f = 0; f < NF; ++f) bfr1[f] = loadB(f, kc + 32);
    }
#pragma unroll
    for (int a = 0; a < MF; ++a)
#pragma unroll
      for (int b = 0; b < NF; ++b)
        acc[a][b] = __builtin_amdgcn_wmma_f32_16x16x32_bf16(
            false, afr0[a], false, bfr0[b], (short)0, acc[a][b], false, false);
    if (!more1) break;

    const bool more2 = (kc + 64) < K;
    if (more2) {
#pragma unroll
      for (int f = 0; f < MF; ++f) afr0[f] = loadA(f, kc + 64);
#pragma unroll
      for (int f = 0; f < NF; ++f) bfr0[f] = loadB(f, kc + 64);
    }
#pragma unroll
    for (int a = 0; a < MF; ++a)
#pragma unroll
      for (int b = 0; b < NF; ++b)
        acc[a][b] = __builtin_amdgcn_wmma_f32_16x16x32_bf16(
            false, afr1[a], false, bfr1[b], (short)0, acc[a][b], false, false);
  }

  // epilogue
#pragma unroll
  for (int a = 0; a < MF; ++a) {
#pragma unroll
    for (int b = 0; b < NF; ++b) {
      int n = n0 + b * 16 + lr;
#pragma unroll
      for (int i = 0; i < 8; ++i) {
        int m = m0 + a * 16 + i + 8 * hi;
        float v = acc[a][b][i];
        if constexpr (OMODE == 0) {
          ((unsigned short*)outp)[(size_t)m * Ntot + n] = f2bf(v + bias[n]);
        } else if constexpr (OMODE == 1) {
          int bb = m >> (2 * ihshift);
          int oy = (((m >> ihshift) & (IH - 1)) << 1) + ph;
          int ox = ((m & (IH - 1)) << 1) + pw;
          size_t idx = (((size_t)(bb * (2 * IH) + oy)) * (2 * IH) + ox) * Ntot + n;
          ((unsigned short*)outp)[idx] = f2bf(v + bias[n]);
        } else {
          if (n < 3) {
            int bb = m >> (2 * ihshift);
            int oy = (((m >> ihshift) & (IH - 1)) << 1) + ph;
            int ox = ((m & (IH - 1)) << 1) + pw;
            ((float*)outp)[(((size_t)(bb * 3 + n)) * 128 + oy) * 128 + ox] = v + bias[n];
          }
        }
      }
    }
  }
}

// ---------------------------------------------------------------------------
// VQ kernel: per wave, 16 rows of z (K=128 per group) resident; sweep 512
// codes with WMMA dots; argmax(2*dot - ||c||^2) == argmin dist; gather code.
// Codebook fragments are double-buffered so loads overlap WMMA + argmax VALU.
// ---------------------------------------------------------------------------
__global__ void __launch_bounds__(256)
vq_k(const unsigned short* __restrict__ ze,   // M x 256 bf16, ch-last
     const unsigned short* __restrict__ cb,   // [2][512][128] bf16
     const float* __restrict__ cnorm,         // [2][512]
     unsigned short* __restrict__ zq) {       // M x 256 bf16
  const int tid  = threadIdx.x;
  const int wave = tid >> 5;
  const int lane = tid & 31;
  const int lr   = lane & 15;
  const int hi   = lane >> 4;
  const int m0   = (blockIdx.x * 8 + wave) * 16;

  for (int g = 0; g < 2; ++g) {
    const unsigned short* zrow = ze + (size_t)(m0 + lr) * 256 + g * 128;
    v16bf afr[4];
#pragma unroll
    for (int c = 0; c < 4; ++c)
      afr[c] = mk16(ld8(zrow + c * 32 + hi * 8), ld8(zrow + c * 32 + 16 + hi * 8));

    const unsigned short* cbg = cb + (size_t)g * 512 * 128;
    float bestv[8]; int besti[8];
#pragma unroll
    for (int i = 0; i < 8; ++i) { bestv[i] = -3.0e38f; besti[i] = 0; }

    auto loadCB = [&](int nt, v16bf* dst) {
      const unsigned short* crow = cbg + (size_t)(nt * 16 + lr) * 128;
#pragma unroll
      for (int c = 0; c < 4; ++c)
        dst[c] = mk16(ld8(crow + c * 32 + hi * 8), ld8(crow + c * 32 + 16 + hi * 8));
    };
    auto score = [&](int nt, const v16bf* bfr) {
      v8f accv;
#pragma unroll
      for (int i = 0; i < 8; ++i) accv[i] = 0.f;
#pragma unroll
      for (int c = 0; c < 4; ++c)
        accv = __builtin_amdgcn_wmma_f32_16x16x32_bf16(
            false, afr[c], false, bfr[c], (short)0, accv, false, false);
      int n = nt * 16 + lr;
      float cn = cnorm[g * 512 + n];
#pragma unroll
      for (int i = 0; i < 8; ++i) {
        float s = 2.f * accv[i] - cn;
        if (s > bestv[i]) { bestv[i] = s; besti[i] = n; }
      }
    };

    v16bf b0[4], b1[4];
    loadCB(0, b0);
    for (int nt = 0; nt < 32; nt += 2) {
      loadCB(nt + 1, b1);                 // overlaps with score(nt)
      score(nt, b0);
      if (nt + 2 < 32) loadCB(nt + 2, b0); // overlaps with score(nt+1)
      score(nt + 1, b1);
    }

    // butterfly max-reduce across the 16 lanes holding different columns
#pragma unroll
    for (int mask = 1; mask < 16; mask <<= 1) {
#pragma unroll
      for (int i = 0; i < 8; ++i) {
        float ov = __shfl_xor(bestv[i], mask, 32);
        int   oi = __shfl_xor(besti[i], mask, 32);
        if (ov > bestv[i] || (ov == bestv[i] && oi < besti[i])) { bestv[i] = ov; besti[i] = oi; }
      }
    }
    // gather winning codebook rows (zq == q in forward); 16 lanes x 8 elems
#pragma unroll
    for (int i = 0; i < 8; ++i) {
      int m = m0 + i + 8 * hi;
      v8bf q = ld8(cbg + (size_t)besti[i] * 128 + lr * 8);
      *(v8bf*)(zq + (size_t)m * 256 + g * 128 + lr * 8) = q;
    }
  }
}

// ---------------------------------------------------------------------------
// Launch: needs ~165 MB workspace (buffers aliased across pipeline stages).
// ---------------------------------------------------------------------------
extern "C" void kernel_launch(void* const* d_in, const int* in_sizes, int n_in,
                              void* d_out, int out_size, void* d_ws, size_t ws_size,
                              hipStream_t stream) {
  (void)in_sizes; (void)n_in; (void)out_size; (void)ws_size;
  const float* x      = (const float*)d_in[0];
  const float* enc0_w = (const float*)d_in[1];
  const float* enc0_b = (const float*)d_in[2];
  const float* cb0    = (const float*)d_in[3];
  const float* dec0_w = (const float*)d_in[4];
  const float* dec0_b = (const float*)d_in[5];
  const float* enc1_w = (const float*)d_in[6];
  const float* enc1_b = (const float*)d_in[7];
  const float* cb1    = (const float*)d_in[8];
  const float* dec1_w = (const float*)d_in[9];
  const float* dec1_b = (const float*)d_in[10];
  float* out = (float*)d_out;

  char* ws = (char*)d_ws;
  size_t off = 0;
  auto take = [&](size_t n) {
    char* p = ws + off;
    off = (off + n + 255) & ~(size_t)255;
    return p;
  };
  unsigned short* ze0   = (unsigned short*)take(131072UL * 256 * 2); // z_e0
  unsigned short* zq0h1 = (unsigned short*)take(131072UL * 256 * 2); // z_q0, later h1
  unsigned short* a0ze1 = (unsigned short*)take(131072UL * 64 * 2);  // im2col A0, later z_e1
  unsigned short* zq1   = (unsigned short*)take(32768UL * 256 * 2);  // z_q1
  unsigned short* wp0   = (unsigned short*)take(256UL * 64 * 2);
  unsigned short* wp1   = (unsigned short*)take(256UL * 4096 * 2);
  unsigned short* wp1d  = (unsigned short*)take(4UL * 256 * 1024 * 2);
  unsigned short* wp0d  = (unsigned short*)take(4UL * 16 * 1024 * 2);
  unsigned short* cbp   = (unsigned short*)take(4UL * 512 * 128 * 2);
  float*          cnorm = (float*)take(4UL * 512 * 4);

  // --- packing ---
  pack_wenc0_k  <<<64,    256, 0, stream>>>(enc0_w, wp0);
  pack_im2col0_k<<<32768, 256, 0, stream>>>(x, a0ze1);
  pack_cb_k     <<<8,     256, 0, stream>>>(cb0, cb1, cbp, cnorm);
  pack_wenc1_k  <<<4096,  256, 0, stream>>>(enc1_w, wp1);
  pack_wdec1_k  <<<4096,  256, 0, stream>>>(dec1_w, wp1d);
  pack_wdec0_k  <<<256,   256, 0, stream>>>(dec0_w, wp0d);

  // --- enc0: A0[131072x64] x Wp0^T -> ze0 (bf16 ch-last) ---
  gemm_k<0, 0, 2, 4, 4, 4><<<dim3(1024, 1), 256, 0, stream>>>(
      a0ze1, wp0, enc0_b, ze0, 256, 64, 0, 0, 0);

  // --- VQ0: ze0 -> zq0 ---
  vq_k<<<1024, 256, 0, stream>>>(ze0, cbp, cnorm, zq0h1);

  // --- enc1: implicit conv GEMM, zq0 (B,64,64,256) -> ze1 (32768x256) ---
  gemm_k<1, 0, 2, 4, 4, 4><<<dim3(256, 1), 256, 0, stream>>>(
      zq0h1, wp1, enc1_b, a0ze1, 256, 4096, 0, 0, 0);

  // --- VQ1: ze1 -> zq1 ---
  vq_k<<<256, 256, 0, stream>>>(a0ze1, cbp + 2UL * 512 * 128, cnorm + 2 * 512, zq1);

  // --- dec1: convT, 4 parity classes, zq1 (B,32,32,256) -> h1 (B,64,64,256) ---
  for (int pc = 0; pc < 4; ++pc)
    gemm_k<2, 1, 2, 4, 4, 4><<<dim3(256, 1), 256, 0, stream>>>(
        zq1, wp1d + (size_t)pc * 256 * 1024, dec1_b, zq0h1,
        256, 1024, pc >> 1, pc & 1, 5);

  // --- dec0: convT, 4 parity classes, h1 (B,64,64,256) -> recon f32 NCHW ---
  for (int pc = 0; pc < 4; ++pc)
    gemm_k<2, 2, 8, 1, 4, 1><<<dim3(256, 1), 256, 0, stream>>>(
        zq0h1, wp0d + (size_t)pc * 16 * 1024, dec0_b, out,
        16, 1024, pc >> 1, pc & 1, 6);
}
